// Entropy_70136815944310
// MI455X (gfx1250) — compile-verified, matrix operand order checked by reading, not run on hardware
//
#include <hip/hip_runtime.h>
#include <hip/hip_bf16.h>

typedef __attribute__((ext_vector_type(16))) _Float16 v16h;
typedef __attribute__((ext_vector_type(8)))  float    v8f;

#define TILE   16
#define HALO   20           // TILE + 4 (5x5 window, SAME padding)
#define NSLOT  13           // ceil(400/32) slots per lane
#define NBINS  256
#define HWDIM  96
#define TPP    36           // tiles per plane: 6x6

#define LN2F      0.6931471805599453f
#define LOG2EF    1.4426950408889634f

__global__ __launch_bounds__(32)
void Entropy_70136815944310_kernel(const float* __restrict__ x,
                                   float* __restrict__ out) {
    // per-wave LDS staging; kbuf padded to 13*32 so phase 1 needs NO guards
    __shared__ float kbuf[NSLOT * 32];                  // 400 live + 16 pad
    __shared__ __align__(16) _Float16 ksumT[16 * 32];   // [col][K], K padded to 32

    const int lane  = threadIdx.x;                      // 0..31 (wave32)
    const int tile  = blockIdx.x;                       // 0..863
    const int plane = tile / TPP;                       // (b,c) flattened, 0..23
    const int t     = tile % TPP;
    const int tt    = (t / 6) * TILE;                   // tile top  (H)
    const int tl    = (t % 6) * TILE;                   // tile left (W)

    const float* xp = x + plane * (HWDIM * HWDIM);

    // ---- cache this lane's share of the 20x20 halo (13 slots cover 400+pad) ----
    float xreg[NSLOT];
#pragma unroll
    for (int s = 0; s < NSLOT; ++s) {
        int i = s * 32 + lane;
        float v = -1.0e9f;                  // sentinel: k underflows to exactly 0
        if (i < HALO * HALO) {
            int r = i / HALO, c = i % HALO;
            int h = tt - 2 + r, w = tl - 2 + c;
            if (h >= 0 && h < HWDIM && w >= 0 && w < HWDIM) v = xp[h * HWDIM + w];
        }
        xreg[s] = v;
    }

    // ---- zero the padded K rows (20..31) of B staging: 0 x NaN would poison WMMA ----
    for (int s = lane; s < 16 * 12; s += 32) {
        int col = s / 12, K = 20 + (s % 12);
        ksumT[col * 32 + K] = (_Float16)0.0f;
    }

    // ---- constant A fragment: 16x32 f16 band matrix, A[m][k] = (k in [m, m+4]) ----
    // 16-bit A layout: lane holds M = lane&15; half h = lane>>4 selects K groups:
    // vgprs 0..3 -> K = 8h + 0..7 ; vgprs 4..7 -> K = 16 + 8h + 0..7
    const int m  = lane & 15;
    const int hh = lane >> 4;
    v16h a;
#pragma unroll
    for (int e = 0; e < 16; ++e) {
        int K = (e < 8) ? (8 * hh + e) : (16 + 8 * hh + (e - 8));
        a[e] = (_Float16)((K >= m && K <= m + 4) ? 1.0f : 0.0f);
    }

    float Zacc[8], Eacc[8];
#pragma unroll
    for (int i = 0; i < 8; ++i) { Zacc[i] = 0.0f; Eacc[i] = 0.0f; }

    __syncthreads();

    for (int b = 0; b < NBINS; ++b) {
        const float bf = (float)b;

        // phase 1 (branch-free): k = e^{-|d|}/(1+e^{-|d|})^2, d = (x-b)/0.1
        // exp(-|d|) = exp2(-|x-b| * 10*log2(e)); raw v_exp_f32, arg always finite
        // pad slots carry the sentinel -> k = 0, stored into the pad region
#pragma unroll
        for (int s = 0; s < NSLOT; ++s) {
            float ad = fabsf(xreg[s] - bf) * (10.0f * LOG2EF);
            float e  = __builtin_amdgcn_exp2f(-ad);
            float r  = __builtin_amdgcn_rcpf(1.0f + e);   // 1+e in [1,2]
            kbuf[s * 32 + lane] = e * r * r;
        }
        __syncthreads();

        // phase 2: horizontal 5-sum -> ksumT[col][K]  (320 tasks, 10 per lane)
#pragma unroll
        for (int s = 0; s < 10; ++s) {
            int tsk = s * 32 + lane;
            int col = tsk & 15;          // output column 0..15
            int K   = tsk >> 4;          // halo row 0..19
            const float* kr = &kbuf[K * HALO + col];
            float sum = kr[0] + kr[1] + kr[2] + kr[3] + kr[4];
            ksumT[col * 32 + K] = (_Float16)sum;
        }
        __syncthreads();

        // phase 3: vertical 5-sum as one banded WMMA: P(16x16) = A(16x32) x B(32x16)
        // B layout mirror of A: lane holds N = lane&15, K groups by half hh.
        union { float4 f4[2]; v16h v; } bu;
        const float4* bp = (const float4*)&ksumT[(lane & 15) * 32 + 8 * hh];
        bu.f4[0] = bp[0];                // f16 K = 8h .. 8h+7
        bu.f4[1] = bp[2];                // f16 K = 16+8h .. 16+8h+7
        v8f c = {};
        v8f d = __builtin_amdgcn_wmma_f32_16x16x32_f16(
                    false, a, false, bu.v, (short)0, c, false, false);

        // entropy accumulation: Z += P ; E += P * ln(P + eps)
        // P + 1e-10 is always a normal f32 -> raw v_log_f32 (log2) * ln2
#pragma unroll
        for (int i = 0; i < 8; ++i) {
            float P  = d[i];
            float lp = __builtin_amdgcn_logf(P + 1e-10f) * LN2F;
            Zacc[i] += P;
            Eacc[i] += P * lp;
        }
        __syncthreads();
    }

    // H = (Z/Zt)*ln(Zt) - E/Zt,  Zt = Z + eps   (== -sum p*ln(p+eps), p = P/Zt)
    float* op = out + plane * (HWDIM * HWDIM);
#pragma unroll
    for (int i = 0; i < 8; ++i) {
        float Zt  = Zacc[i] + 1e-10f;
        float lz  = __builtin_amdgcn_logf(Zt) * LN2F;
        float rz  = __builtin_amdgcn_rcpf(Zt);
        float ent = (Zacc[i] * rz) * lz - Eacc[i] * rz;
        int hrow = tt + i + 8 * hh;          // C/D layout: M = vgpr + 8*(lane/16)
        int wcol = tl + (lane & 15);         //             N = lane % 16
        op[hrow * HWDIM + wcol] = ent;
    }
}

extern "C" void kernel_launch(void* const* d_in, const int* in_sizes, int n_in,
                              void* d_out, int out_size, void* d_ws, size_t ws_size,
                              hipStream_t stream) {
    (void)in_sizes; (void)n_in; (void)d_ws; (void)ws_size; (void)out_size;
    const float* x = (const float*)d_in[0];
    float* out = (float*)d_out;
    // 24 planes * 6x6 tiles of 16x16 = 864 single-wave workgroups
    dim3 grid(24 * TPP), block(32);
    Entropy_70136815944310_kernel<<<grid, block, 0, stream>>>(x, out);
}
